// VQModule_16192026705965
// MI455X (gfx1250) — matrix operand order repaired and checked by reading, not compiled, not run on hardware
//
#include <hip/hip_runtime.h>
#include <hip/hip_bf16.h>

// ---- problem constants ----
#define N_ROWS   65536      // 4 * 128 * 128 unfolded positions
#define EDIM     512
#define NE       1024
#define RVQ_DEPTH 6
#define FOLD_ELEMS 33554432 // 4*128*256*256
#define LOSS_OFF  FOLD_ELEMS
#define IDX_OFF  (FOLD_ELEMS + 1)

// LDS strides (padded for bank-conflict-free WMMA fragment loads)
#define SF4 129             // fp32 row stride in float4s (516 floats)
#define SB  520             // bf16 row stride (halfs): 1040B -> lanes land on distinct banks

typedef __attribute__((ext_vector_type(16))) __bf16 v16bf;
typedef __attribute__((ext_vector_type(8)))  __bf16 v8bf;
typedef __attribute__((ext_vector_type(4)))  __bf16 v4bf;
typedef __attribute__((ext_vector_type(8)))  float  v8f;

// Global (addrspace 1) pointer types: keep provenance through the launder so the
// codebook streams as global_load_b128 (not flat_load, which also ties up DScnt).
typedef const __bf16 __attribute__((address_space(1)))* gbf_ptr;
typedef const v16bf  __attribute__((address_space(1)))* gv16bf_ptr;

// ---------- prologue: bf16 codebook + ||e||^2 ----------
__global__ __launch_bounds__(256) void vq_prep_kernel(const float* __restrict__ emb,
                                                      unsigned short* __restrict__ embbf_u,
                                                      float* __restrict__ esq) {
    __shared__ float sr[256];
    __bf16* embbf = reinterpret_cast<__bf16*>(embbf_u);
    int r = blockIdx.x, t = threadIdx.x;
    float a = emb[r * EDIM + t];
    float b = emb[r * EDIM + t + 256];
    embbf[r * EDIM + t]       = (__bf16)a;
    embbf[r * EDIM + t + 256] = (__bf16)b;
    sr[t] = a * a + b * b;
    __syncthreads();
    for (int s = 128; s > 0; s >>= 1) { if (t < s) sr[t] += sr[t + s]; __syncthreads(); }
    if (t == 0) esq[r] = sr[0];
}

// ---------- fused residual-VQ main kernel ----------
// One 256-thread (8-wave) block owns 16 rows of zf for all 6 depth iterations.
__global__ __launch_bounds__(256) void vq_rvq_kernel(const float* __restrict__ x,
                                                     const float* __restrict__ emb,
                                                     const unsigned short* __restrict__ embbf_u,
                                                     const float* __restrict__ esq,
                                                     float* __restrict__ out,
                                                     float* __restrict__ lossp) {
    const __bf16* embbf = reinterpret_cast<const __bf16*>(embbf_u);

    __shared__ float4 s_rf[16 * SF4];  // residual, fp32 master
    __shared__ float4 s_zq[16 * SF4];  // z_q accumulator
    __shared__ __bf16 s_bf[16 * SB];   // residual, bf16 (WMMA A operand)
    __shared__ float  s_redv[8][16];
    __shared__ int    s_redi[8][16];
    __shared__ int    s_fin[16];
    __shared__ float  s_l[256];

    const int tid  = threadIdx.x;
    const int lane = tid & 31;
    const int wv   = tid >> 5;          // wave id 0..7 -> codes [wv*128, wv*128+128)
    const int m    = lane & 15;         // lane within half
    const int hi   = lane >> 4;         // lane half

    const int nBase = blockIdx.x * 16;
    const int row   = tid >> 4;         // 0..15, 16 threads per row
    const int esub  = tid & 15;         // this thread owns channels c = esub + 16*i

    // ---- unfold: zf row = nBase+row; e = c*4 + ki*2 + kj ----
    const int n   = nBase + row;
    const int bb  = n >> 14;            // / (128*128)
    const int rem = n & 16383;
    const int hb  = rem >> 7;
    const int wb  = rem & 127;

    // base of the 2x2 spatial block for channel 0 of this row
    const float* xblk = x + ((size_t)(bb * 128) * 256 + 2 * hb) * 256 + 2 * wb;

#pragma unroll
    for (int i = 0; i < 8; ++i) {
        int c = esub + (i << 4);
        float2 a0 = *reinterpret_cast<const float2*>(xblk + (size_t)c * 65536);
        float2 a1 = *reinterpret_cast<const float2*>(xblk + (size_t)c * 65536 + 256);
        s_rf[row * SF4 + c] = make_float4(a0.x, a0.y, a1.x, a1.y);
        s_zq[row * SF4 + c] = make_float4(0.f, 0.f, 0.f, 0.f);
    }

    // per-tile ||e||^2, constant over depth
    float e2t[8];
#pragma unroll
    for (int t = 0; t < 8; ++t) e2t[t] = esq[wv * 128 + t * 16 + m];

    // Per-lane fragment base pointers (constant); all tile offsets are immediates.
    const __bf16* abase0 = &s_bf[m * SB + (hi << 4)];                      // A: K half at hi*16
    gbf_ptr bbase0 = (gbf_ptr)(embbf + (wv * 128 + m) * EDIM + (hi << 4)); // B: col=code, K half

    for (int d = 0; d < RVQ_DEPTH; ++d) {
        __syncthreads();
        // refresh bf16 copy of residual (4 values per ds_store_b64)
#pragma unroll
        for (int i = 0; i < 8; ++i) {
            int c = esub + (i << 4);
            float4 v = s_rf[row * SF4 + c];
            v4bf h;
            h.x = (__bf16)v.x; h.y = (__bf16)v.y; h.z = (__bf16)v.z; h.w = (__bf16)v.w;
            *reinterpret_cast<v4bf*>(&s_bf[row * SB + 4 * c]) = h;
        }
        __syncthreads();

        // ---- GEMM: dot(resid_tile[16xK], emb[codes]^T) via wmma bf16 ----
        v8f acc[8] = {};
#pragma unroll 1
        for (int kk = 0; kk < 16; ++kk) {
            // Launder each iteration: forbids hoisting the depth-invariant codebook
            // loads out of the depth loop (would spill); AS(1) keeps global_load.
            gbf_ptr bb_p = bbase0;
            asm volatile("" : "+v"(bb_p));

            // A fragment (contiguous K order): halves at K = kk*32 + {hi*16, hi*16+8}
            union { v16bf v; v8bf h[2]; } au;
            au.h[0] = *reinterpret_cast<const v8bf*>(abase0 + (kk << 5));
            au.h[1] = *reinterpret_cast<const v8bf*>(abase0 + (kk << 5) + 8);
            v16bf a = au.v;

            // Batch all 8 B fragments, then issue 8 back-to-back WMMAs.
            v16bf bm[8];
#pragma unroll
            for (int t = 0; t < 8; ++t)
                bm[t] = *(gv16bf_ptr)(bb_p + t * 16 * EDIM + (kk << 5));
#pragma unroll
            for (int t = 0; t < 8; ++t)
                acc[t] = __builtin_amdgcn_wmma_f32_16x16x32_bf16(
                    false, a, false, bm[t], (short)0, acc[t], false, false);
        }

        // ---- score = ||e||^2 - 2*dot ; branchless argmin per row ----
#pragma unroll
        for (int g = 0; g < 8; ++g) {
            float bv = 3.4e38f; int bi = 0;
#pragma unroll
            for (int t = 0; t < 8; ++t) {
                float s  = e2t[t] - 2.0f * acc[t][g];
                int   ci = wv * 128 + t * 16 + m;
                bool better = (s < bv) | ((s == bv) & (ci < bi));
                bv = better ? s : bv;
                bi = better ? ci : bi;
            }
#pragma unroll
            for (int msk = 1; msk < 16; msk <<= 1) {
                float ov = __shfl_xor(bv, msk, 32);
                int   oi = __shfl_xor(bi, msk, 32);
                bool better = (ov < bv) | ((ov == bv) & (oi < bi));
                bv = better ? ov : bv;
                bi = better ? oi : bi;
            }
            if (m == 0) {                       // lane 0 -> row g, lane 16 -> row g+8
                int rr = g + (hi << 3);
                s_redv[wv][rr] = bv;
                s_redi[wv][rr] = bi;
            }
        }
        __syncthreads();

        if (tid < 16) {                          // combine 8 waves, first-index tie-break
            float bv = s_redv[0][tid]; int bi = s_redi[0][tid];
#pragma unroll
            for (int w = 1; w < 8; ++w) {
                float ov = s_redv[w][tid]; int oi = s_redi[w][tid];
                bool better = (ov < bv) | ((ov == bv) & (oi < bi));
                bv = better ? ov : bv;
                bi = better ? oi : bi;
            }
            s_fin[tid] = bi;
            out[IDX_OFF + d * N_ROWS + nBase + tid] = (float)bi;
        }
        __syncthreads();

        // ---- residual -= emb[idx]; z_q += emb[idx] (float4 granularity) ----
        int code = s_fin[row];
#pragma unroll
        for (int i = 0; i < 8; ++i) {
            int c = esub + (i << 4);
            float4 ev = *reinterpret_cast<const float4*>(emb + (size_t)code * EDIM + 4 * c);
            float4 r = s_rf[row * SF4 + c];
            float4 q = s_zq[row * SF4 + c];
            r.x -= ev.x; r.y -= ev.y; r.z -= ev.z; r.w -= ev.w;
            q.x += ev.x; q.y += ev.y; q.z += ev.z; q.w += ev.w;
            s_rf[row * SF4 + c] = r;
            s_zq[row * SF4 + c] = q;
        }
    }
    __syncthreads();

    // ---- fold z_q out (float2 stores) + per-block loss partial (sum resid^2) ----
    float* oblk = out + ((size_t)(bb * 128) * 256 + 2 * hb) * 256 + 2 * wb;
    float ls = 0.0f;
#pragma unroll
    for (int i = 0; i < 8; ++i) {
        int c = esub + (i << 4);
        float4 rv = s_rf[row * SF4 + c];
        ls += rv.x * rv.x + rv.y * rv.y + rv.z * rv.z + rv.w * rv.w;
        float4 q = s_zq[row * SF4 + c];
        *reinterpret_cast<float2*>(oblk + (size_t)c * 65536)       = make_float2(q.x, q.y);
        *reinterpret_cast<float2*>(oblk + (size_t)c * 65536 + 256) = make_float2(q.z, q.w);
    }
    s_l[tid] = ls;
    __syncthreads();
    for (int s = 128; s > 0; s >>= 1) { if (tid < s) s_l[tid] += s_l[tid + s]; __syncthreads(); }
    if (tid == 0) lossp[blockIdx.x] = s_l[0];
}

// ---------- epilogue: deterministic loss reduction ----------
__global__ __launch_bounds__(256) void vq_loss_kernel(const float* __restrict__ lossp,
                                                      float* __restrict__ out) {
    __shared__ float sr[256];
    int t = threadIdx.x;
    float s = 0.0f;
    for (int i = t; i < N_ROWS / 16; i += 256) s += lossp[i];
    sr[t] = s;
    __syncthreads();
    for (int k = 128; k > 0; k >>= 1) { if (t < k) sr[t] += sr[t + k]; __syncthreads(); }
    // loss = (1 + beta) * mean(resid^2) = 1.25 * sum / (N * EDIM)
    if (t == 0) out[LOSS_OFF] = sr[0] * (1.25f / 33554432.0f);
}

extern "C" void kernel_launch(void* const* d_in, const int* in_sizes, int n_in,
                              void* d_out, int out_size, void* d_ws, size_t ws_size,
                              hipStream_t stream) {
    const float* x   = (const float*)d_in[0];   // [4,128,256,256] fp32
    const float* emb = (const float*)d_in[1];   // [1024,512] fp32
    float* out = (float*)d_out;

    char* ws = (char*)d_ws;
    unsigned short* embbf = (unsigned short*)ws;                 // 1 MiB bf16 codebook
    float* esq   = (float*)(ws + (1u << 20));                    // 4 KiB ||e||^2
    float* lossp = (float*)(ws + (1u << 20) + 4096);             // 16 KiB loss partials

    vq_prep_kernel<<<dim3(NE), dim3(256), 0, stream>>>(emb, embbf, esq);
    vq_rvq_kernel<<<dim3(N_ROWS / 16), dim3(256), 0, stream>>>(x, emb, embbf, esq, out, lossp);
    vq_loss_kernel<<<dim3(1), dim3(256), 0, stream>>>(lossp, out);
}
